// RNNLayer_80650895884768
// MI455X (gfx1250) — matrix operand order repaired and checked by reading, not compiled, not run on hardware
//
#include <hip/hip_runtime.h>
#include <hip/hip_bf16.h>
#include <math.h>

// ---------------------------------------------------------------------------
// RNN layer for MI455X (gfx1250): bf16 WMMA (v_wmma_f32_16x16x32_bf16) with
// fp32 accumulation, wave32, 2x2 (32x32) register tile per wave, TDM
// (tensor_load_to_lds) staging of the recurrent state, and transposed
// K-pair-packed weights so every fragment load is a b128.
// ---------------------------------------------------------------------------

typedef __attribute__((ext_vector_type(16))) __bf16       v16bf;
typedef __attribute__((ext_vector_type(8)))  float        v8f;
typedef __attribute__((ext_vector_type(8)))  unsigned int v8u;
typedef __attribute__((ext_vector_type(4)))  unsigned int u32x4;
typedef __attribute__((ext_vector_type(8)))  int          i32x8;
typedef __attribute__((ext_vector_type(4)))  int          i32x4;

static constexpr int kI = 512;   // INPUT
static constexpr int kH = 1024;  // HIDDEN
static constexpr int kO = 512;   // OUTPUT
static constexpr int kB = 64;    // batch
static constexpr int kT = 512;   // timesteps

// ---- bf16 helpers (round-to-nearest-even) ---------------------------------
__device__ __forceinline__ unsigned int f2bf_bits(float f) {
    union { float f; unsigned int u; } v; v.f = f;
    unsigned int r = 0x7FFFu + ((v.u >> 16) & 1u);
    return (v.u + r) >> 16;
}
__device__ __forceinline__ unsigned short f2bf(float f) {
    return (unsigned short)f2bf_bits(f);
}
__device__ __forceinline__ float bf2f(unsigned short b) {
    union { unsigned int u; float f; } v; v.u = ((unsigned int)b) << 16;
    return v.f;
}
__device__ __forceinline__ unsigned int pack_bf16(float lo, float hi) {
    return f2bf_bits(lo) | (f2bf_bits(hi) << 16);
}

// ---- WMMA fragment loads (ISA 7.12.2 layouts) ------------------------------
// A fragment, 16x32 bf16, row-major bf16 source [*, lda], K contiguous.
// Lane L: M = mBase + (L&15); K-half = (L>>4)*8; VGPR v(0..3): K-pair
// kHalf+2v, VGPR v+4: K-pair kHalf+16+2v.  Two b128 loads per fragment.
__device__ __forceinline__ v8u load_a_bf16(const unsigned short* A, int lda,
                                           int mBase, int kBase, int lane) {
    const unsigned short* row = A + (size_t)(mBase + (lane & 15)) * lda
                                  + kBase + ((lane >> 4) << 3);
    v8u r;
#pragma unroll
    for (int v = 0; v < 4; ++v) {
        r[v]     = *(const unsigned int*)(row + 2 * v);
        r[v + 4] = *(const unsigned int*)(row + 16 + 2 * v);
    }
    return r;
}

// B fragment, 32x16 bf16 from TRANSPOSED K-pair-packed weights:
// WpT[n * (K/2) + k2] holds bf16 pair (2*k2, 2*k2+1) of column n, so each
// lane's 8 fragment dwords are contiguous (two b128 loads).
// Lane L: N = nBase + (L&15), K-half = (L>>4)*8; dwords 0..3: K-pairs
// kHalf/2+v, dwords 4..7: K-pairs 8+kHalf/2+v.
__device__ __forceinline__ v8u load_b_packedT(const unsigned int* WpT, int K,
                                              int nBase, int kBase, int lane) {
    const unsigned int* p = WpT + (size_t)(nBase + (lane & 15)) * (K >> 1)
                               + (kBase >> 1) + ((lane >> 4) << 2);
    v8u r;
#pragma unroll
    for (int v = 0; v < 4; ++v) { r[v] = p[v]; r[v + 4] = p[v + 8]; }
    return r;
}

__device__ __forceinline__ v8f wmma_bf16(v8u a, v8u b, v8f c) {
    return __builtin_amdgcn_wmma_f32_16x16x32_bf16(
        false, __builtin_bit_cast(v16bf, a),
        false, __builtin_bit_cast(v16bf, b),
        (short)0, c, false, false);
}

// ---- TDM: 2D tile load global->LDS (ISA ch.8 descriptor) -------------------
// Loads tile [rows x cols] of 2-byte elements from a row-major tensor with
// row stride `stride` (elements) at `gaddr` into LDS byte offset `lds_off`.
__device__ __forceinline__ void tdm_load_2d_b16(unsigned long long gaddr,
                                                unsigned int lds_off,
                                                unsigned int cols,
                                                unsigned int rows,
                                                unsigned int stride) {
    u32x4 g0;
    g0[0] = 1u;                                       // count=1, user mode
    g0[1] = lds_off;                                  // LDS byte address
    g0[2] = (unsigned int)gaddr;                      // global addr [31:0]
    g0[3] = (unsigned int)((gaddr >> 32) & 0x1FFFFFFu) | (2u << 30); // type=2
    i32x8 g1;
    g1[0] = (int)(1u << 16);                          // data_size=1 (2 bytes)
    g1[1] = (int)((cols & 0xFFFFu) << 16);            // tensor_dim0[15:0]
    g1[2] = (int)(((cols >> 16) & 0xFFFFu) | ((rows & 0xFFFFu) << 16)); // dim1
    g1[3] = (int)((cols & 0xFFFFu) << 16);            // tile_dim0
    g1[4] = (int)(rows & 0xFFFFu);                    // tile_dim1 (tile_dim2=0)
    g1[5] = (int)stride;                              // tensor_dim0_stride lo
    g1[6] = 0;                                        // stride hi, dim1_stride lo
    g1[7] = 0;
    i32x4 gz = {0, 0, 0, 0};
#if defined(__clang_major__) && __clang_major__ >= 23
    i32x8 gz8 = {0, 0, 0, 0, 0, 0, 0, 0};
    __builtin_amdgcn_tensor_load_to_lds(g0, g1, gz, gz, gz8, 0);
#else
    __builtin_amdgcn_tensor_load_to_lds(g0, g1, gz, gz, 0);
#endif
}

// ---------------------------------------------------------------------------
// Pack fp32 weight [K, N] into transposed K-pair-packed bf16 dwords:
// WpT[n * (K/2) + k2] = pack(W[2*k2][n], W[2*k2+1][n]).
__global__ void pack_w_bf16_t(const float* __restrict__ W,
                              unsigned int* __restrict__ WpT, int K, int N) {
    int idx = blockIdx.x * blockDim.x + threadIdx.x;
    int K2 = K >> 1;
    int total = K2 * N;
    if (idx >= total) return;
    int n = idx / K2, k2 = idx - n * K2;
    WpT[idx] = pack_bf16(W[(size_t)(2 * k2) * N + n],
                         W[(size_t)(2 * k2 + 1) * N + n]);
}

// Pairwise fp32 -> packed bf16 (row-major, K contiguous: plain linear pack).
__global__ void pack_x_bf16(const float* __restrict__ x,
                            unsigned int* __restrict__ xp, int n2) {
    int i = blockIdx.x * blockDim.x + threadIdx.x;
    if (i < n2) xp[i] = pack_bf16(x[(size_t)2 * i], x[(size_t)2 * i + 1]);
}

__global__ void zero_u32(unsigned int* __restrict__ p, int n) {
    int i = blockIdx.x * blockDim.x + threadIdx.x;
    if (i < n) p[i] = 0u;
}

// ---------------------------------------------------------------------------
// Phase 1: xW[m, n] = x[m, :] @ W_xh[:, n], M = B*T (m = b*T + t), K = INPUT.
// One wave per 32x32 tile (2x2 wmma blocking). Output bf16 [M, H].
__global__ void xw_gemm(const unsigned short* __restrict__ xb,
                        const unsigned int* __restrict__ Wxh_p,
                        unsigned short* __restrict__ xw) {
    const int N = kH, K = kI;
    int wave = (blockIdx.x * blockDim.x + threadIdx.x) >> 5;
    int lane = threadIdx.x & 31;
    int tn = wave % (N / 32);
    int tm = wave / (N / 32);

    v8f a00 = {}, a01 = {}, a10 = {}, a11 = {};
    for (int kb = 0; kb < K; kb += 32) {
        v8u A0 = load_a_bf16(xb, K, tm * 32, kb, lane);
        v8u A1 = load_a_bf16(xb, K, tm * 32 + 16, kb, lane);
        v8u B0 = load_b_packedT(Wxh_p, K, tn * 32, kb, lane);
        v8u B1 = load_b_packedT(Wxh_p, K, tn * 32 + 16, kb, lane);
        a00 = wmma_bf16(A0, B0, a00);
        a01 = wmma_bf16(A0, B1, a01);
        a10 = wmma_bf16(A1, B0, a10);
        a11 = wmma_bf16(A1, B1, a11);
    }
    int nc = lane & 15, mc = (lane >> 4) << 3;
#pragma unroll
    for (int s = 0; s < 4; ++s) {
        const v8f& acc = (s == 0) ? a00 : (s == 1) ? a01 : (s == 2) ? a10 : a11;
        int n  = tn * 32 + ((s & 1) << 4) + nc;
        int mb = tm * 32 + ((s >> 1) << 4) + mc;
#pragma unroll
        for (int v = 0; v < 8; ++v)
            xw[(size_t)(mb + v) * N + n] = f2bf(acc[v]);
    }
}

// ---------------------------------------------------------------------------
// Phase 2 step t: h_cur = tanh(h_prev @ W_hh + xW[:, t, :] + b_h).
// Grid: 8 blocks x 256 threads. Block handles 32 batch rows x 256 hidden
// cols; the 32x1024 bf16 A-tile (64 KB) is staged into LDS once via TDM and
// shared by all 8 waves (each wave: one 32x32 output tile, 2x2 wmma).
__global__ void rnn_step(const unsigned short* __restrict__ h_prev,
                         const unsigned int* __restrict__ Whh_p,
                         const unsigned short* __restrict__ xw,
                         const float* __restrict__ b_h,
                         unsigned short* __restrict__ h_cur, int t) {
    __shared__ unsigned short s_h[32 * kH];  // 64 KB

    int mg = blockIdx.x & 1;            // which 32-row group of the batch
    int nChunk = blockIdx.x >> 1;       // 0..3, each = 8 N-groups of 32
    int w = threadIdx.x >> 5;           // wave 0..7
    int lane = threadIdx.x & 31;
    int nBase = (nChunk * 8 + w) * 32;

    if (threadIdx.x == 0) {
        typedef __attribute__((address_space(3))) unsigned short lds_us;
        unsigned int lds_off =
            (unsigned int)(unsigned long long)(lds_us*)&s_h[0];
        tdm_load_2d_b16((unsigned long long)(h_prev + (size_t)mg * 32 * kH),
                        lds_off, /*cols=*/kH, /*rows=*/32, /*stride=*/kH);
    }
    __builtin_amdgcn_s_wait_tensorcnt(0);
    __syncthreads();

    v8f a00 = {}, a01 = {}, a10 = {}, a11 = {};
    for (int kb = 0; kb < kH; kb += 32) {
        v8u A0 = load_a_bf16(s_h, kH, 0, kb, lane);
        v8u A1 = load_a_bf16(s_h, kH, 16, kb, lane);
        v8u B0 = load_b_packedT(Whh_p, kH, nBase, kb, lane);
        v8u B1 = load_b_packedT(Whh_p, kH, nBase + 16, kb, lane);
        a00 = wmma_bf16(A0, B0, a00);
        a01 = wmma_bf16(A0, B1, a01);
        a10 = wmma_bf16(A1, B0, a10);
        a11 = wmma_bf16(A1, B1, a11);
    }
    int nc = lane & 15, mc = (lane >> 4) << 3;
#pragma unroll
    for (int s = 0; s < 4; ++s) {
        const v8f& acc = (s == 0) ? a00 : (s == 1) ? a01 : (s == 2) ? a10 : a11;
        int n  = nBase + ((s & 1) << 4) + nc;
        int mb = mg * 32 + ((s >> 1) << 4) + mc;
        float bias = b_h[n];
#pragma unroll
        for (int v = 0; v < 8; ++v) {
            int m = mb + v;  // batch index
            float val = acc[v] + bias + bf2f(xw[((size_t)m * kT + t) * kH + n]);
            h_cur[(size_t)m * kH + n] = f2bf(tanhf(val));
        }
    }
}

// ---------------------------------------------------------------------------
// Phase 3: y = h_all @ W_hy + b_y over all timesteps at once.
// h_all bf16 [T*B, H], row m = t*B + b; out fp32 [B, T, O].
__global__ void y_gemm(const unsigned short* __restrict__ h_all,
                       const unsigned int* __restrict__ Why_p,
                       const float* __restrict__ b_y,
                       float* __restrict__ out) {
    const int N = kO, K = kH;
    int wave = (blockIdx.x * blockDim.x + threadIdx.x) >> 5;
    int lane = threadIdx.x & 31;
    int tn = wave % (N / 32);
    int tm = wave / (N / 32);

    v8f a00 = {}, a01 = {}, a10 = {}, a11 = {};
    for (int kb = 0; kb < K; kb += 32) {
        v8u A0 = load_a_bf16(h_all, K, tm * 32, kb, lane);
        v8u A1 = load_a_bf16(h_all, K, tm * 32 + 16, kb, lane);
        v8u B0 = load_b_packedT(Why_p, K, tn * 32, kb, lane);
        v8u B1 = load_b_packedT(Why_p, K, tn * 32 + 16, kb, lane);
        a00 = wmma_bf16(A0, B0, a00);
        a01 = wmma_bf16(A0, B1, a01);
        a10 = wmma_bf16(A1, B0, a10);
        a11 = wmma_bf16(A1, B1, a11);
    }
    int nc = lane & 15, mc = (lane >> 4) << 3;
#pragma unroll
    for (int s = 0; s < 4; ++s) {
        const v8f& acc = (s == 0) ? a00 : (s == 1) ? a01 : (s == 2) ? a10 : a11;
        int n  = tn * 32 + ((s & 1) << 4) + nc;
        int mb = tm * 32 + ((s >> 1) << 4) + mc;
        float bias = b_y[n];
#pragma unroll
        for (int v = 0; v < 8; ++v) {
            int m = mb + v;          // m = t*B + b
            int b = m & (kB - 1);
            int tt = m >> 6;         // kB == 64
            out[((size_t)b * kT + tt) * kO + n] = acc[v] + bias;
        }
    }
}

// ---------------------------------------------------------------------------
extern "C" void kernel_launch(void* const* d_in, const int* in_sizes, int n_in,
                              void* d_out, int out_size, void* d_ws, size_t ws_size,
                              hipStream_t stream) {
    const float* x    = (const float*)d_in[0];  // [B, T, I]
    const float* W_xh = (const float*)d_in[1];  // [I, H]
    const float* W_hh = (const float*)d_in[2];  // [H, H]
    const float* b_h  = (const float*)d_in[3];  // [H]
    const float* W_hy = (const float*)d_in[4];  // [H, O]
    const float* b_y  = (const float*)d_in[5];  // [O]
    float* out = (float*)d_out;                 // [B, T, O]

    // Workspace: Wxh_p 1MB | Whh_p 2MB | Why_p 1MB | x_bf16 32MB |
    //            xw 64MB | h_hist ~67MB   (total ~167MB)
    char* p = (char*)d_ws;
    unsigned int*   Wxh_p  = (unsigned int*)p;   p += (size_t)(kI / 2) * kH * 4;
    unsigned int*   Whh_p  = (unsigned int*)p;   p += (size_t)(kH / 2) * kH * 4;
    unsigned int*   Why_p  = (unsigned int*)p;   p += (size_t)(kH / 2) * kO * 4;
    unsigned short* x_bf   = (unsigned short*)p; p += (size_t)kB * kT * kI * 2;
    unsigned short* xw     = (unsigned short*)p; p += (size_t)kB * kT * kH * 2;
    unsigned short* h_hist = (unsigned short*)p; p += (size_t)(kT + 1) * kB * kH * 2;

    const int TB = 256;
    auto cdiv = [](int a, int b) { return (a + b - 1) / b; };

    // Pack weights (transposed K-pair bf16) and convert x to bf16.
    pack_w_bf16_t<<<cdiv((kI / 2) * kH, TB), TB, 0, stream>>>(W_xh, Wxh_p, kI, kH);
    pack_w_bf16_t<<<cdiv((kH / 2) * kH, TB), TB, 0, stream>>>(W_hh, Whh_p, kH, kH);
    pack_w_bf16_t<<<cdiv((kH / 2) * kO, TB), TB, 0, stream>>>(W_hy, Why_p, kH, kO);
    pack_x_bf16<<<cdiv(kB * kT * kI / 2, TB), TB, 0, stream>>>(
        x, (unsigned int*)x_bf, kB * kT * kI / 2);

    // h_0 = 0.
    zero_u32<<<cdiv(kB * kH / 2, TB), TB, 0, stream>>>((unsigned int*)h_hist,
                                                       kB * kH / 2);

    // Phase 1: input projection GEMM (32x32 tile per wave).
    {
        int waves = (kB * kT / 32) * (kH / 32);      // 32768 waves
        xw_gemm<<<waves * 32 / TB, TB, 0, stream>>>(x_bf, Wxh_p, xw);
    }

    // Phase 2: sequential scan (512 dependent launches).
    for (int t = 0; t < kT; ++t) {
        rnn_step<<<8, TB, 0, stream>>>(
            h_hist + (size_t)t * kB * kH, Whh_p, xw, b_h,
            h_hist + (size_t)(t + 1) * kB * kH, t);
    }

    // Phase 3: output projection GEMM over all timesteps.
    {
        int waves = (kB * kT / 32) * (kO / 32);      // 16384 waves
        y_gemm<<<waves * 32 / TB, TB, 0, stream>>>(h_hist + (size_t)kB * kH,
                                                   Why_p, b_y, out);
    }
}
